// VectorQuantizer_85392539779409
// MI455X (gfx1250) — matrix operand order repaired and checked by reading, not compiled, not run on hardware
//
#include <hip/hip_runtime.h>

// Problem constants (fixed by reference setup_inputs)
constexpr int kB = 8;
constexpr int kD = 256;
constexpr int kT = 2048;
constexpr int kK = 8192;
constexpr int kN = kB * kT;            // 16384 rows of z_flat
constexpr int MT = 64;                 // rows per workgroup in main kernel
constexpr int NCHUNK = 64;             // codes staged per LDS chunk
constexpr int NCHUNKS = kK / NCHUNK;   // 128 chunks
constexpr int CHUNK_U4 = NCHUNK * kD / 8;  // 2048 uint4 per chunk

typedef __attribute__((ext_vector_type(16))) __bf16 v16bf;
typedef __attribute__((ext_vector_type(8)))  float  v8f;
typedef int v4i __attribute__((ext_vector_type(4)));

struct Frag32 { uint4 lo; uint4 hi; };  // 32 bytes == v16bf

// ---- async global->LDS support (CDNA5), with sync fallback --------------------
#if defined(__has_builtin)
#if __has_builtin(__builtin_amdgcn_global_load_async_to_lds_b128)
#define HAS_ASYNC_LDS 1
#endif
#endif
#ifndef HAS_ASYNC_LDS
#define HAS_ASYNC_LDS 0
#endif

#if HAS_ASYNC_LDS
typedef __attribute__((address_space(1))) v4i gv4i;   // global (AS1, 64-bit ptr)
typedef __attribute__((address_space(3))) v4i lv4i;   // LDS    (AS3, 32-bit ptr)
__device__ __forceinline__ void async_cp16(const void* g, const void* l) {
  __builtin_amdgcn_global_load_async_to_lds_b128(
      (gv4i*)(uintptr_t)g,                 // 64-bit global pointer
      (lv4i*)(unsigned)(uintptr_t)l,       // 32-bit LDS offset (aperture truncation)
      0, 0);
}
template <int N> __device__ __forceinline__ void wait_async() {
#if __has_builtin(__builtin_amdgcn_s_wait_asynccnt)
  __builtin_amdgcn_s_wait_asynccnt((short)N);
#else
  asm volatile("s_wait_asynccnt %0" ::"i"(N));
#endif
}
#endif

__device__ __forceinline__ unsigned short f2bf(float f) {
  unsigned u = __builtin_bit_cast(unsigned, f);
  unsigned rnd = 0x7fffu + ((u >> 16) & 1u);   // round-to-nearest-even
  return (unsigned short)((u + rnd) >> 16);
}
__device__ __forceinline__ float bf2f(unsigned short h) {
  return __builtin_bit_cast(float, ((unsigned)h) << 16);
}

// ---------------------------------------------------------------------------
// Kernel 1: z (B, D, T) f32  ->  zbf (N=B*T, D) row-major bf16 (tiled transpose)
// ---------------------------------------------------------------------------
__global__ __launch_bounds__(256)
void z_to_bf16_kernel(const float* __restrict__ z, unsigned short* __restrict__ zbf) {
  __shared__ float tile[64][65];
  const int tbase = blockIdx.x * 64;
  const int dbase = blockIdx.y * 64;
  const int b     = blockIdx.z;
  const int c = threadIdx.x & 63;
  const int r = threadIdx.x >> 6;          // 0..3
  const float* zp = z + (size_t)b * kD * kT;
#pragma unroll
  for (int i = 0; i < 16; ++i) {
    const int dl = r * 16 + i;
    tile[dl][c] = zp[(size_t)(dbase + dl) * kT + tbase + c];   // coalesced in t
  }
  __syncthreads();
#pragma unroll
  for (int i = 0; i < 16; ++i) {
    const int tl = r * 16 + i;
    const float v = tile[c][tl];
    zbf[(size_t)(b * kT + tbase + tl) * kD + dbase + c] = f2bf(v);  // coalesced in d
  }
}

// ---------------------------------------------------------------------------
// Kernel 2: codebook (K, D) f32 -> cbf bf16 + cnorm[k] = sum_d bf16(c)^2
// ---------------------------------------------------------------------------
__global__ __launch_bounds__(256)
void cb_prep_kernel(const float* __restrict__ cb, unsigned short* __restrict__ cbf,
                    float* __restrict__ cnorm) {
  const int k = blockIdx.x;
  const int d = threadIdx.x;
  const float v = cb[(size_t)k * kD + d];
  const unsigned short hb = f2bf(v);
  cbf[(size_t)k * kD + d] = hb;
  const float vr = bf2f(hb);
  float s = vr * vr;
#pragma unroll
  for (int m = 1; m < 32; m <<= 1) s += __shfl_xor(s, m, 32);
  __shared__ float ws8[8];
  if ((threadIdx.x & 31) == 0) ws8[threadIdx.x >> 5] = s;
  __syncthreads();
  if (threadIdx.x == 0) {
    float t = 0.f;
#pragma unroll
    for (int w = 0; w < 8; ++w) t += ws8[w];
    cnorm[k] = t;
  }
}

// ---------------------------------------------------------------------------
// Kernel 3: fused bf16-WMMA GEMM + argmin over K, double-buffered async LDS.
//   Grid: N/MT workgroups of 256 threads (8 waves).
//   Wave (mg = w>>1, cw = w&1): rows [mg*16, +16), codes [cw*32, +32) of a chunk.
//   LDS layout: Bs[2][64 rows x 512B], 16B chunks XOR-swizzled by (row & 15)
//   -> B-fragment reads by lanes n=0..15 hit 16 distinct chunks = all 64 banks.
// ---------------------------------------------------------------------------
__global__ __launch_bounds__(256)
void vq_argmin_kernel(const unsigned short* __restrict__ zbf,
                      const unsigned short* __restrict__ cbf,
                      const float* __restrict__ cnorm,
                      int* __restrict__ idxOut, float* __restrict__ idxOutF) {
  __shared__ __align__(16) unsigned short Bs[2][NCHUNK * 256];   // exactly 64 KB

  const int tid  = threadIdx.x;
  const int lane = tid & 31;
  const int wave = tid >> 5;
  const int mg = wave >> 1;         // row group 0..3
  const int cw = wave & 1;          // code half 0..1
  const int n  = lane & 15;
  const int hi = lane >> 4;         // half-wave selector
  const int row0 = blockIdx.x * MT;

  // --- A fragments (16x32 bf16 per k-step), ISA interleaved layout, kept in VGPRs ---
  // lane: m = row, khalf = hi*8; a[j] = A[m][s*32 + khalf + j + (j>=8 ? 8 : 0)]
  v16bf afrag[8];
  {
    const int m = row0 + mg * 16 + n;
    const unsigned short* ap = zbf + (size_t)m * kD + hi * 8;
#pragma unroll
    for (int s = 0; s < 8; ++s) {
      Frag32 f;
      f.lo = *(const uint4*)(ap + s * 32);        // K = khalf + 0..7
      f.hi = *(const uint4*)(ap + s * 32 + 16);   // K = khalf + 16..23
      afrag[s] = __builtin_bit_cast(v16bf, f);
    }
  }

  float bestd[8];
  int   besti[8];
#pragma unroll
  for (int j = 0; j < 8; ++j) { bestd[j] = 3.4e38f; besti[j] = 0; }

  const uint4* csrc = (const uint4*)cbf;      // chunk c starts at uint4 index c*2048
  const int r0 = cw * 32 + n;                 // tile-0 code row in chunk (r0 & 15 == n)
  const int r1 = r0 + 16;                     // tile-1 code row (r1 & 15 == n)

#if HAS_ASYNC_LDS
  // ---- issue chunk 0 DMA ----
  {
#pragma unroll
    for (int i = 0; i < 8; ++i) {
      const int lin = tid + 256 * i;
      const int rr = lin >> 5, ci = lin & 31;
      async_cp16(csrc + lin, &Bs[0][rr * 256 + ((ci ^ (rr & 15)) << 3)]);
    }
  }
#endif

  for (int c = 0; c < NCHUNKS; ++c) {
    const int buf = c & 1;
#if HAS_ASYNC_LDS
    if (c + 1 < NCHUNKS) {   // overlap next chunk's DMA with this chunk's WMMAs
#pragma unroll
      for (int i = 0; i < 8; ++i) {
        const int lin = tid + 256 * i;
        const int rr = lin >> 5, ci = lin & 31;
        async_cp16(csrc + (size_t)(c + 1) * CHUNK_U4 + lin,
                   &Bs[buf ^ 1][rr * 256 + ((ci ^ (rr & 15)) << 3)]);
      }
      wait_async<8>();       // chunk c's 8 ops done; chunk c+1's may be in flight
    } else {
      wait_async<0>();
    }
    __syncthreads();
#else
    // synchronous fallback: stage via VGPRs (Bs[buf] last read 2 chunks ago)
    {
      uint4 tmp[8];
      const uint4* src = csrc + (size_t)c * CHUNK_U4;
#pragma unroll
      for (int i = 0; i < 8; ++i) tmp[i] = src[tid + 256 * i];
#pragma unroll
      for (int i = 0; i < 8; ++i) {
        const int lin = tid + 256 * i;
        const int rr = lin >> 5, ci = lin & 31;
        *(uint4*)(&Bs[buf][rr * 256 + ((ci ^ (rr & 15)) << 3)]) = tmp[i];
      }
    }
    __syncthreads();
#endif

    // ---- 16 WMMAs on this chunk ----
    v8f acc0 = {}; v8f acc1 = {};
    const unsigned short* base = &Bs[buf][0];
    const unsigned short* rp0 = base + r0 * 256;
    const unsigned short* rp1 = base + r1 * 256;
#pragma unroll
    for (int s = 0; s < 8; ++s) {
      const int cb0 = s * 4 + hi;              // 16B-chunk index within the row
      Frag32 f0, f1;
      f0.lo = *(const uint4*)(rp0 + (((cb0    ) ^ n) << 3));
      f0.hi = *(const uint4*)(rp0 + (((cb0 + 2) ^ n) << 3));
      f1.lo = *(const uint4*)(rp1 + (((cb0    ) ^ n) << 3));
      f1.hi = *(const uint4*)(rp1 + (((cb0 + 2) ^ n) << 3));
      const v16bf vb0 = __builtin_bit_cast(v16bf, f0);
      const v16bf vb1 = __builtin_bit_cast(v16bf, f1);
      acc0 = __builtin_amdgcn_wmma_f32_16x16x32_bf16(false, afrag[s], false, vb0,
                                                     (short)0, acc0, false, false);
      acc1 = __builtin_amdgcn_wmma_f32_16x16x32_bf16(false, afrag[s], false, vb1,
                                                     (short)0, acc1, false, false);
    }

    // dist = ||c||^2 - 2*dot  (||z||^2 constant per row -> dropped). k ascending,
    // strict '<' keeps first occurrence (matches jnp.argmin).
    {
      const int k0 = c * NCHUNK + cw * 32 + n;
      const float cn0 = cnorm[k0];
      const float cn1 = cnorm[k0 + 16];
#pragma unroll
      for (int j = 0; j < 8; ++j) {
        const float d0 = fmaf(-2.f, acc0[j], cn0);
        if (d0 < bestd[j]) { bestd[j] = d0; besti[j] = k0; }
        const float d1 = fmaf(-2.f, acc1[j], cn1);
        if (d1 < bestd[j]) { bestd[j] = d1; besti[j] = k0 + 16; }
      }
    }
    __syncthreads();   // all waves done reading Bs[buf] before it is re-filled
  }

  // Cross-lane argmin within each 16-lane half (rows hi*8+j, min over n coverage)
#pragma unroll
  for (int j = 0; j < 8; ++j) {
    float d = bestd[j]; int i = besti[j];
#pragma unroll
    for (int m = 1; m <= 8; m <<= 1) {
      const float od = __shfl_xor(d, m, 32);
      const int   oi = __shfl_xor(i, m, 32);
      if (od < d || (od == d && oi < i)) { d = od; i = oi; }
    }
    bestd[j] = d; besti[j] = i;
  }

  // Cross-wave combine via LDS scratch aliased over Bs (all LDS reads are done).
  struct RedT { float d; int i; };
  RedT* Red = (RedT*)&Bs[0][0];        // [4 row groups][2 code halves][16 rows]
  if (n == 0) {   // lanes 0 (rows 0..7) and 16 (rows 8..15)
#pragma unroll
    for (int j = 0; j < 8; ++j) {
      Red[(mg * 2 + cw) * 16 + hi * 8 + j].d = bestd[j];
      Red[(mg * 2 + cw) * 16 + hi * 8 + j].i = besti[j];
    }
  }
  __syncthreads();
  if (tid < MT) {  // combine the two code-half waves, write final index
    const int r = tid;
    const RedT a = Red[((r >> 4) * 2 + 0) * 16 + (r & 15)];
    const RedT b = Red[((r >> 4) * 2 + 1) * 16 + (r & 15)];
    const int ii = (b.d < a.d || (b.d == a.d && b.i < a.i)) ? b.i : a.i;
    idxOut[row0 + r]  = ii;
    idxOutF[row0 + r] = (float)ii;
  }
}

// ---------------------------------------------------------------------------
// Kernel 4: gather z_q into (B, D, T) + per-block partial sum of (z_q - z)^2
// Grid: (T/16, B). Thread: tt = tid>>4 picks t, dd = tid&15 strides over d.
// ---------------------------------------------------------------------------
__global__ __launch_bounds__(256)
void gather_loss_kernel(const float* __restrict__ z, const float* __restrict__ cb,
                        const int* __restrict__ idx, float* __restrict__ zq,
                        float* __restrict__ partial) {
  const int b = blockIdx.y;
  const int t = blockIdx.x * 16 + (threadIdx.x >> 4);
  const int dd = threadIdx.x & 15;
  const int k = idx[b * kT + t];
  const float* crow = cb + (size_t)k * kD;
  const float* zp = z  + (size_t)b * kD * kT + t;
  float*       qp = zq + (size_t)b * kD * kT + t;
  float s = 0.f;
#pragma unroll
  for (int dg = 0; dg < 16; ++dg) {
    const int d = dg * 16 + dd;
    const float q = crow[d];
    const float diff = q - zp[(size_t)d * kT];
    s += diff * diff;
    qp[(size_t)d * kT] = q;
  }
#pragma unroll
  for (int m = 1; m < 32; m <<= 1) s += __shfl_xor(s, m, 32);
  __shared__ float ws8[8];
  if ((threadIdx.x & 31) == 0) ws8[threadIdx.x >> 5] = s;
  __syncthreads();
  if (threadIdx.x == 0) {
    float tsum = 0.f;
#pragma unroll
    for (int w = 0; w < 8; ++w) tsum += ws8[w];
    partial[blockIdx.y * gridDim.x + blockIdx.x] = tsum;  // deterministic, no atomics
  }
}

// ---------------------------------------------------------------------------
// Kernel 5: reduce 1024 partials -> loss = 1.1 * sum / (B*D*T)
// ---------------------------------------------------------------------------
__global__ __launch_bounds__(256)
void finalize_kernel(const float* __restrict__ partial, float* __restrict__ lossOut) {
  float s = 0.f;
  for (int i = threadIdx.x; i < 1024; i += 256) s += partial[i];
#pragma unroll
  for (int m = 1; m < 32; m <<= 1) s += __shfl_xor(s, m, 32);
  __shared__ float ws8[8];
  if ((threadIdx.x & 31) == 0) ws8[threadIdx.x >> 5] = s;
  __syncthreads();
  if (threadIdx.x == 0) {
    float t = 0.f;
#pragma unroll
    for (int w = 0; w < 8; ++w) t += ws8[w];
    *lossOut = 1.1f * t / (float)((size_t)kB * kD * kT);
  }
}

// ---------------------------------------------------------------------------
extern "C" void kernel_launch(void* const* d_in, const int* in_sizes, int n_in,
                              void* d_out, int out_size, void* d_ws, size_t ws_size,
                              hipStream_t stream) {
  (void)in_sizes; (void)n_in; (void)out_size; (void)ws_size;
  const float* z  = (const float*)d_in[0];   // (B, D, T) f32
  const float* cb = (const float*)d_in[1];   // (K, D)    f32

  // d_out tuple layout: z_q_st (B*D*T) | indices (B*T) as float | loss (1)
  float* out     = (float*)d_out;
  float* zqOut   = out;
  float* idxF    = out + (size_t)kB * kD * kT;
  float* lossOut = idxF + (size_t)kB * kT;

  // Workspace layout
  char* ws = (char*)d_ws;
  unsigned short* zbf   = (unsigned short*)ws;                         // 8 MB
  unsigned short* cbf   = (unsigned short*)(ws + (8u << 20));          // 4 MB
  float*          cnorm = (float*)(ws + (12u << 20));                  // 32 KB
  int*            idxI  = (int*)  (ws + (12u << 20) + (64u << 10));    // 64 KB
  float*          part  = (float*)(ws + (12u << 20) + (128u << 10));   // 4 KB

  z_to_bf16_kernel<<<dim3(kT / 64, kD / 64, kB), 256, 0, stream>>>(z, zbf);
  cb_prep_kernel<<<kK, 256, 0, stream>>>(cb, cbf, cnorm);
  vq_argmin_kernel<<<kN / MT, 256, 0, stream>>>(zbf, cbf, cnorm, idxI, idxF);
  gather_loss_kernel<<<dim3(kT / 16, kB), 256, 0, stream>>>(z, cb, idxI, zqOut, part);
  finalize_kernel<<<1, 256, 0, stream>>>(part, lossOut);
}
